// FBPINN_23459111371265
// MI455X (gfx1250) — matrix operand order repaired and checked by reading, not compiled, not run on hardware
//
#include <hip/hip_runtime.h>
#include <math.h>

typedef __attribute__((ext_vector_type(16))) _Float16 v16h;
typedef __attribute__((ext_vector_type(8)))  float    v8f;

#define NSUB   16
#define HW     32
#define TPB    256
#define WAVES  (TPB / 32)
#define PTSW   16            // points per wave (WMMA N dimension)
#define PI_F   3.14159265358979f

// CDNA5 hardware TANH transcendental.
__device__ __forceinline__ float fast_tanh(float v) {
#if __has_builtin(__builtin_amdgcn_tanhf)
  return __builtin_amdgcn_tanhf(v);
#else
  return tanhf(v);
#endif
}

// Fixed neuron permutation baked into layer-2 A columns so that layer-1's
// D layout feeds layer-2's B operand with zero cross-lane traffic:
//   D (lane half h) holds neurons {8h+j} (tile0) and {16+8h+j} (tile1);
//   B slot s at half h must be input neuron nu(s+16h).
__device__ __forceinline__ int nu(int c) {
  return (c < 8) ? c : (c < 16) ? c + 8 : (c < 24) ? c - 8 : c;
}

__global__ __launch_bounds__(TPB) void fbpinn_wmma(
    const float* __restrict__ x,
    const float* __restrict__ W0, const float* __restrict__ b0,
    const float* __restrict__ W1, const float* __restrict__ b1,
    const float* __restrict__ W2, const float* __restrict__ b2,
    const float* __restrict__ W3, const float* __restrict__ b3,
    float* __restrict__ out, int n)
{
  // f16 A-operand image of W1/W2 in the CDNA5 16-bit 16x32 A layout:
  //   [subnet][layer][tile][lane][slot] = 64 KB.
  // Slot s -> K position c = s + 16*(lane>=16) (this code's contiguous k order, matched by B).
  // Layer 2 additionally permutes columns: value = W2[row][nu(c)].
  __shared__ __align__(32) _Float16 aimg[NSUB * 2 * 2 * 32 * 16];
  // Small params staged as f32 for contiguous 32B LDS reads (10.3 KB).
  __shared__ __align__(32) float w0s[NSUB * HW];
  __shared__ __align__(32) float b0s[NSUB * HW];
  __shared__ __align__(32) float b1s[NSUB * HW];
  __shared__ __align__(32) float b2s[NSUB * HW];
  __shared__ __align__(32) float w3s[NSUB * HW];
  __shared__ float b3s[NSUB];

  const int tid = threadIdx.x;
  for (int e = tid; e < NSUB * 2 * 2 * 32 * 16; e += TPB) {
    int s     = e & 15;
    int lane_ = (e >> 4) & 31;
    int tile  = (e >> 9) & 1;
    int layer = (e >> 10) & 1;
    int k     = e >> 11;
    int row   = tile * 16 + (lane_ & 15);            // output neuron (A row)
    int c     = s + 16 * (lane_ >> 4);               // K position
    int col   = layer ? nu(c) : c;                   // input neuron
    const float* Wsrc = layer ? W2 : W1;
    aimg[e] = (_Float16)Wsrc[(k * HW + row) * HW + col];
  }
  for (int e = tid; e < NSUB * HW; e += TPB) {
    w0s[e] = W0[e];
    b0s[e] = b0[e];
    b1s[e] = b1[e];
    b2s[e] = b2[e];
    w3s[e] = W3[e];
    if ((e & (HW - 1)) == 0) b3s[e >> 5] = b3[e >> 5];
  }
  __syncthreads();

  const int lane = tid & 31;
  const int wave = tid >> 5;
  const int p    = lane & 15;        // point within tile (WMMA N)
  const int half = lane >> 4;
  const int tileBase = (blockIdx.x * WAVES + wave) * PTSW;
  if (tileBase >= n) return;         // uniform per wave: EXEC stays all-ones

  const float xp = x[tileBase + p];
  float part = 0.0f, bias_acc = 0.0f, wsum = 0.0f;
  const v16h* av = (const v16h*)aimg;

  for (int k = 0; k < NSUB; ++k) {
    // ---- partition-of-unity window (TW=0.05, OVERLAP=0.06) ----
    const float xmin = (float)k * (1.0f / 16.0f) - 0.06f;
    const float xmax = (float)(k + 1) * (1.0f / 16.0f) + 0.06f;
    float up = fminf(fmaxf((xp - (xmin - 0.025f)) * 20.0f, 0.0f), 1.0f);
    float dn = fminf(fmaxf(((xmax + 0.025f) - xp) * 20.0f, 0.0f), 1.0f);
    float wk = 0.25f * (1.0f - __cosf(PI_F * up)) * (1.0f - __cosf(PI_F * dn));
    float center = 0.5f * (xmin + xmax);
    float scale  = fmaxf(0.5f * (xmax - xmin), 1e-9f);
    float xn = (xp - center) / scale;

    // ---- layer 0 (32x1): each lane computes exactly the 16 h0 values its
    //      B1 slots need (neurons 16*half .. 16*half+15), straight into f16.
    const v8f* w0v = (const v8f*)&w0s[k * HW + 16 * half];
    const v8f* b0v = (const v8f*)&b0s[k * HW + 16 * half];
    v8f w0a = w0v[0], w0b = w0v[1];
    v8f b0a = b0v[0], b0b = b0v[1];
    v16h bmat;
#pragma unroll
    for (int j = 0; j < 8; ++j) {
      bmat[j]     = (_Float16)fast_tanh(fmaf(w0a[j], xn, b0a[j]));
      bmat[j + 8] = (_Float16)fast_tanh(fmaf(w0b[j], xn, b0b[j]));
    }

    // ---- layer 1: D = W1 x h0 + b1 (rows in identity order) ----
    v8f c0 = *(const v8f*)&b1s[k * HW + 8 * half];
    v8f c1 = *(const v8f*)&b1s[k * HW + 16 + 8 * half];
    v16h a0 = av[((k * 2 + 0) * 2 + 0) * 32 + lane];
    v16h a1 = av[((k * 2 + 0) * 2 + 1) * 32 + lane];
    v8f d0 = __builtin_amdgcn_wmma_f32_16x16x32_f16(
        false, a0, false, bmat, (short)0, c0, false, false);
    v8f d1 = __builtin_amdgcn_wmma_f32_16x16x32_f16(
        false, a1, false, bmat, (short)0, c1, false, false);

    // ---- activation + in-lane repack (nu-permuted columns: no shuffles) ----
#pragma unroll
    for (int j = 0; j < 8; ++j) {
      bmat[j]     = (_Float16)fast_tanh(d0[j]);
      bmat[j + 8] = (_Float16)fast_tanh(d1[j]);
    }

    // ---- layer 2: D = W2[:, nu] x h1' + b2 ----
    c0 = *(const v8f*)&b2s[k * HW + 8 * half];
    c1 = *(const v8f*)&b2s[k * HW + 16 + 8 * half];
    a0 = av[((k * 2 + 1) * 2 + 0) * 32 + lane];
    a1 = av[((k * 2 + 1) * 2 + 1) * 32 + lane];
    d0 = __builtin_amdgcn_wmma_f32_16x16x32_f16(
        false, a0, false, bmat, (short)0, c0, false, false);
    d1 = __builtin_amdgcn_wmma_f32_16x16x32_f16(
        false, a1, false, bmat, (short)0, c1, false, false);

    // ---- output layer (1x32 dot): this lane's 16 neurons only; the
    //      partner half is folded in ONCE after the subnet loop. ----
    v8f w3a = *(const v8f*)&w3s[k * HW + 8 * half];
    v8f w3b = *(const v8f*)&w3s[k * HW + 16 + 8 * half];
    float acc = 0.0f;
#pragma unroll
    for (int j = 0; j < 8; ++j) {
      acc = fmaf(w3a[j], fast_tanh(d0[j]), acc);
      acc = fmaf(w3b[j], fast_tanh(d1[j]), acc);
    }
    part     = fmaf(wk, acc, part);
    bias_acc = fmaf(wk, b3s[k], bias_acc);
    wsum    += wk;
  }

  // Partner lane (L^16) accumulated the other 16 neurons with identical wk
  // (same point), so one half-swap completes every subnet's dot product.
  float total = part + __shfl_xor(part, 16, 32) + bias_acc;
  if (lane < 16) out[tileBase + p] = total / (wsum + 1e-12f);
}

extern "C" void kernel_launch(void* const* d_in, const int* in_sizes, int n_in,
                              void* d_out, int out_size, void* d_ws, size_t ws_size,
                              hipStream_t stream) {
  const float* x  = (const float*)d_in[0];
  const float* W0 = (const float*)d_in[1];
  const float* b0 = (const float*)d_in[2];
  const float* W1 = (const float*)d_in[3];
  const float* b1 = (const float*)d_in[4];
  const float* W2 = (const float*)d_in[5];
  const float* b2 = (const float*)d_in[6];
  const float* W3 = (const float*)d_in[7];
  const float* b3 = (const float*)d_in[8];
  float* out = (float*)d_out;
  const int n = in_sizes[0];                       // 524288 points
  const int blocks = (n + WAVES * PTSW - 1) / (WAVES * PTSW);
  fbpinn_wmma<<<blocks, TPB, 0, stream>>>(x, W0, b0, W1, b1, W2, b2, W3, b3,
                                          out, n);
}
